// DecoderRNN_57475252355161
// MI455X (gfx1250) — matrix operand order repaired and checked by reading, not compiled, not run on hardware
//
#include <hip/hip_runtime.h>
#include <hip/hip_bf16.h>
#include <math.h>

#define BB 32       // batch
#define HH 512      // hidden
#define EHH 1024    // 2H encoder dim
#define EE 512      // embedding dim
#define SS 128      // source len
#define TT 64       // target len
#define NSTEP 63    // decode steps
#define KX0 1536    // E + 2H (GRU0 input width)
#define VV 32000    // vocab
#define KFC 1536    // H + 2H (fc input width)
#define MTOT (BB*NSTEP)  // 2016 GEMM rows

typedef __attribute__((ext_vector_type(16))) __bf16 v16bf;
typedef __attribute__((ext_vector_type(8)))  __bf16 v8bf;
typedef __attribute__((ext_vector_type(8)))  float  v8f;
typedef __attribute__((ext_vector_type(4)))  float  f4v;

__device__ __forceinline__ float sigmoidf_(float x) {
  return 1.f / (1.f + __expf(-x));
}

// ---------------------------------------------------------------- init ----
__global__ void init_state(const float* __restrict__ hidden,
                           float* __restrict__ h0f, float* __restrict__ h1f,
                           __bf16* __restrict__ h0b, __bf16* __restrict__ h1b) {
  int i = blockIdx.x * blockDim.x + threadIdx.x;
  if (i < BB * HH) {
    float a = hidden[i];            // hidden[0]
    float b = hidden[BB * HH + i];  // hidden[1]
    h0f[i] = a; h1f[i] = b;
    h0b[i] = (__bf16)a; h1b[i] = (__bf16)b;
  }
}

// ------------------------------------------------ f32 -> bf16 conversion --
__global__ void f32_to_bf16(const float* __restrict__ src,
                            __bf16* __restrict__ dst, int n8) {
  int i = blockIdx.x * blockDim.x + threadIdx.x;
  if (i < n8) {
    f4v a = ((const f4v*)src)[2 * i];
    f4v b = ((const f4v*)src)[2 * i + 1];
    v8bf o;
    for (int e = 0; e < 4; ++e) { o[e] = (__bf16)a[e]; o[4 + e] = (__bf16)b[e]; }
    ((v8bf*)dst)[i] = o;
  }
}

// ------------------------------------------------- attention + embedding --
// one block (1024 thr) per batch row; writes x0bf [32,1536] and ctx into Xbf
__global__ void __launch_bounds__(1024)
attn_step(const int* __restrict__ tgt, const float* __restrict__ h1prev,
          const float* __restrict__ enc, const unsigned char* __restrict__ mask,
          const float* __restrict__ emb_table, const float* __restrict__ Wa,
          __bf16* __restrict__ x0bf, __bf16* __restrict__ Xbf, int t) {
  int b = blockIdx.x;
  int tid = threadIdx.x;
  __shared__ float sh_h[HH];
  __shared__ float sh_q[EHH];
  __shared__ float sh_s[SS];
  __shared__ float sh_part[SS * 8];
  __shared__ float sh_red[2];

  if (tid < HH) sh_h[tid] = h1prev[b * HH + tid];
  __syncthreads();

  // q = h1 @ Wa^T   (Wa: [EH, H] row-major) ; one e per thread
  {
    int e = tid;  // 1024 threads == EHH
    const float* wr = Wa + (size_t)e * HH;
    float acc = 0.f;
    for (int k = 0; k < HH; ++k) acc += sh_h[k] * wr[k];
    sh_q[e] = acc;
  }
  __syncthreads();

  // scores: 8 partial sums per source position
  {
    int s = tid >> 3;         // 0..127
    int p = tid & 7;          // 0..7
    const float* er = enc + ((size_t)b * SS + s) * EHH + p * 128;
    const float* qr = sh_q + p * 128;
    float acc = 0.f;
    for (int k = 0; k < 128; ++k) acc += qr[k] * er[k];
    sh_part[s * 8 + p] = acc;
  }
  __syncthreads();
  if (tid < SS) {
    float acc = 0.f;
    for (int p = 0; p < 8; ++p) acc += sh_part[tid * 8 + p];
    acc *= 0.03125f;  // 1/sqrt(1024)
    if (!mask[b * SS + tid]) acc = -INFINITY;
    sh_s[tid] = acc;
  }
  __syncthreads();
  if (tid == 0) {
    float m = sh_s[0];
    for (int s = 1; s < SS; ++s) m = fmaxf(m, sh_s[s]);
    sh_red[0] = m;
  }
  __syncthreads();
  if (tid < SS) sh_s[tid] = __expf(sh_s[tid] - sh_red[0]);
  __syncthreads();
  if (tid == 0) {
    float sum = 0.f;
    for (int s = 0; s < SS; ++s) sum += sh_s[s];
    sh_red[1] = 1.0f / sum;
  }
  __syncthreads();
  float inv = sh_red[1];

  // ctx = softmax(scores) @ enc  (one e per thread, coalesced over e)
  {
    int e = tid;
    float acc = 0.f;
    for (int s = 0; s < SS; ++s)
      acc += sh_s[s] * enc[((size_t)b * SS + s) * EHH + e];
    acc *= inv;
    __bf16 cb = (__bf16)acc;
    x0bf[b * KX0 + EE + e] = cb;
    Xbf[((size_t)(b * NSTEP + t)) * KFC + HH + e] = cb;
  }
  // embedding gather into x0bf[:, :E]
  if (tid < EE) {
    int tok = tgt[b * TT + t];
    x0bf[b * KX0 + tid] = (__bf16)emb_table[(size_t)tok * EE + tid];
  }
}

// ------------------------------------------------------- WMMA GRU cell ----
// block: 128 thr = 4 waves; wave = {gi,gh} x {batch half}; one 16-col gate
// tile per block (grid.x = 512/16 = 32). gi: G = x @ w_ih^T ; gh: G = h @ w_hh^T.
__global__ void __launch_bounds__(128)
gru_wmma(const __bf16* __restrict__ xbf, int Kx,        // [32, Kx]
         const __bf16* __restrict__ hbf,                // [32, 512] prev h bf16
         const float* __restrict__ hprevf,              // [32, 512] prev h f32
         const __bf16* __restrict__ wihb,               // [1536, Kx]  bf16
         const __bf16* __restrict__ whhb,               // [1536, 512] bf16
         const float* __restrict__ b_ih, const float* __restrict__ b_hh,
         float* __restrict__ hnewf, __bf16* __restrict__ hnewbf,
         __bf16* __restrict__ Xbf, int t) {
  int lane = threadIdx.x & 31;
  int wave = threadIdx.x >> 5;   // 0..3
  int mtile = wave & 1;          // batch rows 0-15 / 16-31
  int role = wave >> 1;          // 0 = gi, 1 = gh
  int j0 = blockIdx.x * 16;
  int lh = lane & 15;
  bool hiHalf = lane >= 16;

  const __bf16* A = role ? hbf : xbf;
  const __bf16* Wt = role ? whhb : wihb;
  int K = role ? HH : Kx;

  v8f acc[3];
  for (int g = 0; g < 3; ++g)
    for (int r = 0; r < 8; ++r) acc[g][r] = 0.f;

  const __bf16* aptr = A + (size_t)(mtile * 16 + lh) * K + (hiHalf ? 8 : 0);
  int boff = hiHalf ? 8 : 0;
  for (int k0 = 0; k0 < K; k0 += 32) {
    v16bf a;
    {
      v8bf lo = *reinterpret_cast<const v8bf*>(aptr + k0);
      v8bf hi = *reinterpret_cast<const v8bf*>(aptr + k0 + 16);
      for (int e = 0; e < 8; ++e) { a[e] = lo[e]; a[8 + e] = hi[e]; }
    }
    for (int g = 0; g < 3; ++g) {
      int n = j0 + g * HH + lh;
      const __bf16* wp = Wt + (size_t)n * K + k0 + boff;
      v16bf bf;
      {
        v8bf lo = *reinterpret_cast<const v8bf*>(wp);
        v8bf hi = *reinterpret_cast<const v8bf*>(wp + 16);
        for (int e = 0; e < 8; ++e) { bf[e] = lo[e]; bf[8 + e] = hi[e]; }
      }
      acc[g] = __builtin_amdgcn_wmma_f32_16x16x32_bf16(
          false, a, false, bf, (short)0, acc[g], false, false);
    }
  }

  __shared__ float gbuf[2][3][BB][16];   // role, gate, batch, col = 12 KB
  for (int g = 0; g < 3; ++g)
    for (int r = 0; r < 8; ++r)
      gbuf[role][g][mtile * 16 + r + (hiHalf ? 8 : 0)][lh] = acc[g][r];
  __syncthreads();

  // fused gate epilogue: 512 outputs, 4 per thread
  for (int idx = threadIdx.x; idx < BB * 16; idx += blockDim.x) {
    int b = idx >> 4;
    int jl = idx & 15;
    int j = j0 + jl;
    float ir = gbuf[0][0][b][jl] + b_ih[j];
    float iz = gbuf[0][1][b][jl] + b_ih[j + HH];
    float in_ = gbuf[0][2][b][jl] + b_ih[j + 2 * HH];
    float hr = gbuf[1][0][b][jl] + b_hh[j];
    float hz = gbuf[1][1][b][jl] + b_hh[j + HH];
    float hn = gbuf[1][2][b][jl] + b_hh[j + 2 * HH];
    float r = sigmoidf_(ir + hr);
    float z = sigmoidf_(iz + hz);
    float n = tanhf(in_ + r * hn);
    float h = (1.f - z) * n + z * hprevf[b * HH + j];
    hnewf[b * HH + j] = h;
    __bf16 hb16 = (__bf16)h;
    hnewbf[b * HH + j] = hb16;
    if (Xbf) Xbf[((size_t)(b * NSTEP + t)) * KFC + j] = hb16;
  }
}

// ------------------------------------------------------- big WMMA GEMM ----
// out[2016, 32000] = Xbf[2016,1536] * W^T + bias.
// Block tile 96x256 (2016 = 21*96, 32000 = 125*256), grid=(21,125).
// 8 waves = 2 M-groups (48 rows) x 4 N-groups (64 cols); each wave:
// 3 A-frags x 4 B-tiles = 12 accumulators. W panel read once per 96 M rows
// -> L2 W traffic = 21 * 98 MB (3x less than a 32-row M block).
template <bool WBF>
__global__ void __launch_bounds__(256)
fc_gemm(const __bf16* __restrict__ Xbf, const float* __restrict__ Wf,
        const __bf16* __restrict__ Wb, const float* __restrict__ bias,
        float* __restrict__ out) {
  int lane = threadIdx.x & 31;
  int wave = threadIdx.x >> 5;         // 0..7
  int wm = wave & 1;                   // M half (48 rows)
  int wn = wave >> 1;                  // 0..3  (64 cols)
  int m0 = blockIdx.x * 96 + wm * 48;
  int nb = blockIdx.y * 256 + wn * 64;
  int lh = lane & 15;
  bool hiHalf = lane >= 16;

  v8f acc[12];                          // [mf*4 + i], mf<3, i<4
  for (int i = 0; i < 4; ++i) {
    float bv = bias[nb + i * 16 + lh];
    for (int mf = 0; mf < 3; ++mf)
      for (int r = 0; r < 8; ++r) acc[mf * 4 + i][r] = bv;
  }

  const __bf16* abase = Xbf + (size_t)(m0 + lh) * KFC + (hiHalf ? 8 : 0);
  int boff = hiHalf ? 8 : 0;

  for (int k0 = 0; k0 < KFC; k0 += 32) {
    v16bf a[3];
    for (int mf = 0; mf < 3; ++mf) {
      const __bf16* ap = abase + (size_t)(16 * mf) * KFC + k0;
      v8bf lo = *reinterpret_cast<const v8bf*>(ap);
      v8bf hi = *reinterpret_cast<const v8bf*>(ap + 16);
      for (int e = 0; e < 8; ++e) { a[mf][e] = lo[e]; a[mf][8 + e] = hi[e]; }
    }
    for (int i = 0; i < 4; ++i) {
      int n = nb + i * 16 + lh;
      v16bf bf;
      if (WBF) {
        const __bf16* wp = Wb + (size_t)n * KFC + k0 + boff;
        v8bf lo = *reinterpret_cast<const v8bf*>(wp);
        v8bf hi = *reinterpret_cast<const v8bf*>(wp + 16);
        for (int e = 0; e < 8; ++e) { bf[e] = lo[e]; bf[8 + e] = hi[e]; }
      } else {
        const float* wp = Wf + (size_t)n * KFC + k0 + boff;
        f4v w0 = *reinterpret_cast<const f4v*>(wp);
        f4v w1 = *reinterpret_cast<const f4v*>(wp + 4);
        f4v w2 = *reinterpret_cast<const f4v*>(wp + 16);
        f4v w3 = *reinterpret_cast<const f4v*>(wp + 20);
        for (int e = 0; e < 4; ++e) {
          bf[e] = (__bf16)w0[e];      bf[4 + e] = (__bf16)w1[e];
          bf[8 + e] = (__bf16)w2[e];  bf[12 + e] = (__bf16)w3[e];
        }
      }
      for (int mf = 0; mf < 3; ++mf)
        acc[mf * 4 + i] = __builtin_amdgcn_wmma_f32_16x16x32_bf16(
            false, a[mf], false, bf, (short)0, acc[mf * 4 + i], false, false);
    }
  }

  for (int mf = 0; mf < 3; ++mf) {
    for (int i = 0; i < 4; ++i) {
      int n = nb + i * 16 + lh;
      for (int r = 0; r < 8; ++r) {
        int mrow = m0 + mf * 16 + r + (hiHalf ? 8 : 0);
        out[(size_t)mrow * VV + n] = acc[mf * 4 + i][r];
      }
    }
  }
}

// ----------------------------------------------------------------- host ---
extern "C" void kernel_launch(void* const* d_in, const int* in_sizes, int n_in,
                              void* d_out, int out_size, void* d_ws, size_t ws_size,
                              hipStream_t stream) {
  (void)in_sizes; (void)n_in; (void)out_size;
  const int*   tgt    = (const int*)d_in[0];
  const float* hidden = (const float*)d_in[1];
  const float* enc    = (const float*)d_in[2];
  const unsigned char* mask = (const unsigned char*)d_in[3];
  const float* emb    = (const float*)d_in[4];
  const float* Wa     = (const float*)d_in[5];
  const float* w_ih0  = (const float*)d_in[6];
  const float* w_hh0  = (const float*)d_in[7];
  const float* b_ih0  = (const float*)d_in[8];
  const float* b_hh0  = (const float*)d_in[9];
  const float* w_ih1  = (const float*)d_in[10];
  const float* w_hh1  = (const float*)d_in[11];
  const float* b_ih1  = (const float*)d_in[12];
  const float* b_hh1  = (const float*)d_in[13];
  const float* fc_w   = (const float*)d_in[14];
  const float* fc_b   = (const float*)d_in[15];
  float* out = (float*)d_out;

  // ---- workspace layout (all chunks multiples of 16 B) ----
  __bf16* Xbf   = (__bf16*)d_ws;                       // 2016*1536
  __bf16* x0bf  = Xbf + (size_t)MTOT * KFC;            // 32*1536
  __bf16* h0bf  = x0bf + BB * KX0;                     // 2 * 32*512
  __bf16* h1bf  = h0bf + 2 * BB * HH;                  // 2 * 32*512
  __bf16* wih0b = h1bf + 2 * BB * HH;                  // 1536*1536
  __bf16* whh0b = wih0b + (size_t)3 * HH * KX0;        // 1536*512
  __bf16* wih1b = whh0b + (size_t)3 * HH * HH;         // 1536*512
  __bf16* whh1b = wih1b + (size_t)3 * HH * HH;         // 1536*512
  float*  h0f   = (float*)(whh1b + (size_t)3 * HH * HH);
  float*  h1f   = h0f + 2 * BB * HH;
  __bf16* fcwb  = (__bf16*)(h1f + 2 * BB * HH);        // optional 32000*1536
  size_t need_base = (size_t)((char*)fcwb - (char*)d_ws);
  bool useWbf = ws_size >= need_base + (size_t)VV * KFC * sizeof(__bf16);

  // ---- one-time (per launch) weight conversions ----
  {
    int n8;
    n8 = 3 * HH * KX0 / 8;
    f32_to_bf16<<<dim3((n8 + 255) / 256), dim3(256), 0, stream>>>(w_ih0, wih0b, n8);
    n8 = 3 * HH * HH / 8;
    f32_to_bf16<<<dim3((n8 + 255) / 256), dim3(256), 0, stream>>>(w_hh0, whh0b, n8);
    f32_to_bf16<<<dim3((n8 + 255) / 256), dim3(256), 0, stream>>>(w_ih1, wih1b, n8);
    f32_to_bf16<<<dim3((n8 + 255) / 256), dim3(256), 0, stream>>>(w_hh1, whh1b, n8);
    if (useWbf) {
      int m8 = VV * KFC / 8;
      f32_to_bf16<<<dim3((m8 + 255) / 256), dim3(256), 0, stream>>>(fc_w, fcwb, m8);
    }
  }
  init_state<<<dim3(64), dim3(256), 0, stream>>>(hidden, h0f, h1f, h0bf, h1bf);

  // ---- sequential decode: attention -> GRU0 -> GRU1, X row as side effect ----
  for (int t = 0; t < NSTEP; ++t) {
    int p = t & 1, q = (t + 1) & 1;
    const float*  h0fp = h0f + p * BB * HH;
    const float*  h1fp = h1f + p * BB * HH;
    float*  h0fn = h0f + q * BB * HH;
    float*  h1fn = h1f + q * BB * HH;
    const __bf16* h0bp = h0bf + p * BB * HH;
    const __bf16* h1bp = h1bf + p * BB * HH;
    __bf16* h0bn = h0bf + q * BB * HH;
    __bf16* h1bn = h1bf + q * BB * HH;

    attn_step<<<dim3(BB), dim3(1024), 0, stream>>>(tgt, h1fp, enc, mask, emb,
                                                   Wa, x0bf, Xbf, t);
    gru_wmma<<<dim3(32), dim3(128), 0, stream>>>(x0bf, KX0, h0bp, h0fp,
                                                 wih0b, whh0b, b_ih0, b_hh0,
                                                 h0fn, h0bn, (__bf16*)nullptr, t);
    gru_wmma<<<dim3(32), dim3(128), 0, stream>>>(h0bn, HH, h1bp, h1fp,
                                                 wih1b, whh1b, b_ih1, b_hh1,
                                                 h1fn, h1bn, Xbf, t);
  }

  // ---- deferred output projection: 2016 x 32000 x 1536 (+bias) ----
  if (useWbf)
    fc_gemm<true><<<dim3(21, 125), dim3(256), 0, stream>>>(Xbf, fc_w, fcwb, fc_b, out);
  else
    fc_gemm<false><<<dim3(21, 125), dim3(256), 0, stream>>>(Xbf, fc_w, fcwb, fc_b, out);
}
